// HLGLayers_8461085573554
// MI455X (gfx1250) — matrix operand order repaired and checked by reading, not compiled, not run on hardware
//
#include <hip/hip_runtime.h>
#include <hip/hip_bf16.h>

typedef __attribute__((ext_vector_type(16))) _Float16 v16h;
typedef __attribute__((ext_vector_type(8)))  float    v8f;

#define GW    7      // window size
#define NTOK  49     // tokens per window
#define NPAD  64     // padded tokens
#define HEADS 8
#define HD    64     // head dim
#define CDIM  512
#define PDIM  32
#define NBIAS 169    // (2G-1)^2

// workspace layout
#define WS_TABLE_OFF 0                  // 169*8 f32
#define WS_BIASF_OFF 8192               // bytes; 8*4*32*32 f32 = 128KB
#define BIASF_COUNT  (HEADS * 4 * 32 * 32)

// ---------------------------------------------------------------------------
// Kernel 1: DynamicPosBias MLP -> table[169][8]
// ---------------------------------------------------------------------------
__device__ __forceinline__ void ln_relu_mm(const float* x, float* y,
                                           const float* g, const float* b,
                                           const float* w, const float* bias,
                                           int outw) {
    float mu = 0.f;
    #pragma unroll
    for (int i = 0; i < PDIM; ++i) mu += x[i];
    mu *= (1.f / PDIM);
    float var = 0.f;
    #pragma unroll
    for (int i = 0; i < PDIM; ++i) { float d = x[i] - mu; var += d * d; }
    var *= (1.f / PDIM);
    float inv = rsqrtf(var + 1e-5f);
    float t[PDIM];
    #pragma unroll
    for (int i = 0; i < PDIM; ++i) {
        float u = (x[i] - mu) * inv * g[i] + b[i];
        t[i] = u > 0.f ? u : 0.f;
    }
    for (int j = 0; j < outw; ++j) {
        float s = bias[j];
        #pragma unroll
        for (int i = 0; i < PDIM; ++i) s += t[i] * w[i * outw + j];
        y[j] = s;
    }
}

__global__ void pos_mlp_kernel(const float* __restrict__ w0, const float* __restrict__ b0,
                               const float* __restrict__ g1, const float* __restrict__ be1,
                               const float* __restrict__ w1, const float* __restrict__ bb1,
                               const float* __restrict__ g2, const float* __restrict__ be2,
                               const float* __restrict__ w2, const float* __restrict__ bb2,
                               const float* __restrict__ g3, const float* __restrict__ be3,
                               const float* __restrict__ w3, const float* __restrict__ bb3,
                               float* __restrict__ table) {
    int r = blockIdx.x * blockDim.x + threadIdx.x;
    if (r >= NBIAS) return;
    float bh = (float)(r / (2 * GW - 1)) - (float)(GW - 1);
    float bw = (float)(r % (2 * GW - 1)) - (float)(GW - 1);
    float x[PDIM], y[PDIM], o[HEADS];
    #pragma unroll
    for (int j = 0; j < PDIM; ++j)
        x[j] = bh * w0[j] + bw * w0[PDIM + j] + b0[j];
    ln_relu_mm(x, y, g1, be1, w1, bb1, PDIM);
    ln_relu_mm(y, x, g2, be2, w2, bb2, PDIM);
    ln_relu_mm(x, o, g3, be3, w3, bb3, HEADS);
    #pragma unroll
    for (int hh = 0; hh < HEADS; ++hh) table[r * HEADS + hh] = o[hh];
}

// ---------------------------------------------------------------------------
// Kernel 1b: expand bias into WMMA C-fragment layout, mask folded in.
// biasF[h][wave][lane][nt*8 + r]  (contiguous 32 floats per lane)
// ---------------------------------------------------------------------------
__global__ void bias_expand_kernel(const float* __restrict__ table,
                                   float* __restrict__ biasF) {
    int i = blockIdx.x * blockDim.x + threadIdx.x;
    if (i >= BIASF_COUNT) return;
    int j    = i & 31;
    int lane = (i >> 5) & 31;
    int wave = (i >> 10) & 3;
    int h    = i >> 12;
    int nt = j >> 3, r = j & 7;
    int n = wave * 16 + ((lane >> 4) << 3) + r;   // query row
    int m = nt * 16 + (lane & 15);                // key col
    float val = -1e30f;
    if (n < NTOK && m < NTOK) {
        int idx = (n / GW - m / GW + GW - 1) * (2 * GW - 1) + (n % GW - m % GW + GW - 1);
        val = table[idx * HEADS + h];
    }
    biasF[i] = val;
}

// ---------------------------------------------------------------------------
// WMMA fragment loaders from LDS (16-bit operands, 16x16x32 shape)
// A 16x32: lane<16 -> row=lane, K {0..7,16..23}; lane>=16 -> K {8..15,24..31}
// B 32x16: lane<16 -> col=lane, K {0..7,16..23}; lane>=16 -> K {8..15,24..31}
// ---------------------------------------------------------------------------
__device__ __forceinline__ v16h load_fragA(const _Float16* base, int row0, int kk,
                                           int lane, int stride) {
    const _Float16* p = base + (row0 + (lane & 15)) * stride + kk * 32 + ((lane >> 4) << 3);
    union { v16h v; uint4 u[2]; } r;
    r.u[0] = *(const uint4*)(p);
    r.u[1] = *(const uint4*)(p + 16);
    return r.v;
}

__device__ __forceinline__ v16h load_fragB(const _Float16* base, int col0, int kk,
                                           int lane, int stride) {
    const _Float16* p = base + (col0 + (lane & 15)) * stride + kk * 32 + ((lane >> 4) << 3);
    union { v16h v; uint4 u[2]; } r;
    r.u[0] = *(const uint4*)(p);
    r.u[1] = *(const uint4*)(p + 16);
    return r.v;
}

__device__ __forceinline__ v8f wmma_f16(v16h a, v16h b, v8f c) {
    return __builtin_amdgcn_wmma_f32_16x16x32_f16(false, a, false, b,
                                                  (short)0, c, false, false);
}

// ---------------------------------------------------------------------------
// Kernel 2: one workgroup (4 wave32) per (batch, head)
// ---------------------------------------------------------------------------
__global__ void __launch_bounds__(128)
attn_kernel(const float* __restrict__ q, const float* __restrict__ k,
            const float* __restrict__ v, const float* __restrict__ biasF,
            float* __restrict__ out) {
    __shared__ __align__(16) _Float16 Qs[NPAD * HD];
    __shared__ __align__(16) _Float16 Ks[NPAD * HD];
    __shared__ __align__(16) _Float16 Vts[HD * NPAD];   // transposed V
    __shared__ __align__(16) _Float16 Ps[NPAD * NPAD];  // softmax probs

    const int bh   = blockIdx.x;
    const int b    = bh >> 3;
    const int h    = bh & 7;
    const int tid  = threadIdx.x;
    const int lane = tid & 31;
    const int wave = tid >> 5;

    // zero-fill padded operand tiles (uint4 stores)
    {
        uint4 z = {0u, 0u, 0u, 0u};
        for (int i = tid; i < NPAD * HD / 8; i += 128) {
            ((uint4*)Qs)[i]  = z;
            ((uint4*)Ks)[i]  = z;
            ((uint4*)Vts)[i] = z;
        }
    }

    // load + f32->f16 convert (attention scale 1/sqrt(64) folded into Q)
    const size_t base = ((size_t)b * NTOK) * CDIM + (size_t)h * HD;
    const float* qb = q + base;
    const float* kb = k + base;
    const float* vb = v + base;
    for (int idx = tid; idx < NTOK * (HD / 4); idx += 128) {
        int n  = idx >> 4;          // 16 float4 per row
        int d4 = idx & 15;
        int d  = d4 << 2;
        float4 fq = ((const float4*)(qb + n * CDIM))[d4];
        float4 fk = ((const float4*)(kb + n * CDIM))[d4];
        float4 fv = ((const float4*)(vb + n * CDIM))[d4];
        union { _Float16 hh[4]; uint2 u; } pq, pk;
        pq.hh[0] = (_Float16)(fq.x * 0.125f); pq.hh[1] = (_Float16)(fq.y * 0.125f);
        pq.hh[2] = (_Float16)(fq.z * 0.125f); pq.hh[3] = (_Float16)(fq.w * 0.125f);
        pk.hh[0] = (_Float16)fk.x; pk.hh[1] = (_Float16)fk.y;
        pk.hh[2] = (_Float16)fk.z; pk.hh[3] = (_Float16)fk.w;
        *(uint2*)(Qs + n * HD + d) = pq.u;
        *(uint2*)(Ks + n * HD + d) = pk.u;
        Vts[(d + 0) * NPAD + n] = (_Float16)fv.x;
        Vts[(d + 1) * NPAD + n] = (_Float16)fv.y;
        Vts[(d + 2) * NPAD + n] = (_Float16)fv.z;
        Vts[(d + 3) * NPAD + n] = (_Float16)fv.w;
    }
    __syncthreads();

    // -------- GEMM 1: S = Q * K^T (strip of 16 rows per wave) --------
    v8f acc0 = {}, acc1 = {}, acc2 = {}, acc3 = {};
    {
        v16h a0 = load_fragA(Qs, wave * 16, 0, lane, HD);
        v16h a1 = load_fragA(Qs, wave * 16, 1, lane, HD);
        v16h b0, b1;
        b0 = load_fragB(Ks,  0, 0, lane, HD); b1 = load_fragB(Ks,  0, 1, lane, HD);
        acc0 = wmma_f16(a0, b0, acc0); acc0 = wmma_f16(a1, b1, acc0);
        b0 = load_fragB(Ks, 16, 0, lane, HD); b1 = load_fragB(Ks, 16, 1, lane, HD);
        acc1 = wmma_f16(a0, b0, acc1); acc1 = wmma_f16(a1, b1, acc1);
        b0 = load_fragB(Ks, 32, 0, lane, HD); b1 = load_fragB(Ks, 32, 1, lane, HD);
        acc2 = wmma_f16(a0, b0, acc2); acc2 = wmma_f16(a1, b1, acc2);
        b0 = load_fragB(Ks, 48, 0, lane, HD); b1 = load_fragB(Ks, 48, 1, lane, HD);
        acc3 = wmma_f16(a0, b0, acc3); acc3 = wmma_f16(a1, b1, acc3);
    }

    // -------- bias add (precomputed fragment layout, mask folded) --------
    const int colL   = lane & 15;
    const int rowOff = (lane >> 4) << 3;
    float sc[4][8];
    #pragma unroll
    for (int r = 0; r < 8; ++r) { sc[0][r] = acc0[r]; sc[1][r] = acc1[r]; sc[2][r] = acc2[r]; sc[3][r] = acc3[r]; }

    {
        const float4* bF = (const float4*)(biasF + (((h * 4 + wave) * 32) + lane) * 32);
        #pragma unroll
        for (int nt = 0; nt < 4; ++nt) {
            float4 blo = bF[nt * 2 + 0];
            float4 bhi = bF[nt * 2 + 1];
            sc[nt][0] += blo.x; sc[nt][1] += blo.y; sc[nt][2] += blo.z; sc[nt][3] += blo.w;
            sc[nt][4] += bhi.x; sc[nt][5] += bhi.y; sc[nt][6] += bhi.z; sc[nt][7] += bhi.w;
        }
    }

    // -------- row softmax (rows live across 16-lane halves) --------
    #pragma unroll
    for (int r = 0; r < 8; ++r) {
        float mx = fmaxf(fmaxf(sc[0][r], sc[1][r]), fmaxf(sc[2][r], sc[3][r]));
        #pragma unroll
        for (int s = 1; s < 16; s <<= 1) mx = fmaxf(mx, __shfl_xor(mx, s, 32));
        float sum = 0.f;
        #pragma unroll
        for (int nt = 0; nt < 4; ++nt) {
            float e = __expf(sc[nt][r] - mx);
            sc[nt][r] = e;
            sum += e;
        }
        #pragma unroll
        for (int s = 1; s < 16; s <<= 1) sum += __shfl_xor(sum, s, 32);
        float inv = 1.f / sum;
        #pragma unroll
        for (int nt = 0; nt < 4; ++nt) sc[nt][r] *= inv;
    }

    // store P to LDS (row-major; masked cols underflowed to 0)
    #pragma unroll
    for (int nt = 0; nt < 4; ++nt) {
        int col = nt * 16 + colL;
        #pragma unroll
        for (int r = 0; r < 8; ++r) {
            int row = wave * 16 + rowOff + r;
            Ps[row * NPAD + col] = (_Float16)sc[nt][r];
        }
    }
    __syncthreads();

    // -------- GEMM 2: O = P * V  (B operand = rows of V^T) --------
    v8f o0 = {}, o1 = {}, o2 = {}, o3 = {};
    {
        v16h a0 = load_fragA(Ps, wave * 16, 0, lane, NPAD);
        v16h a1 = load_fragA(Ps, wave * 16, 1, lane, NPAD);
        v16h b0, b1;
        b0 = load_fragB(Vts,  0, 0, lane, NPAD); b1 = load_fragB(Vts,  0, 1, lane, NPAD);
        o0 = wmma_f16(a0, b0, o0); o0 = wmma_f16(a1, b1, o0);
        b0 = load_fragB(Vts, 16, 0, lane, NPAD); b1 = load_fragB(Vts, 16, 1, lane, NPAD);
        o1 = wmma_f16(a0, b0, o1); o1 = wmma_f16(a1, b1, o1);
        b0 = load_fragB(Vts, 32, 0, lane, NPAD); b1 = load_fragB(Vts, 32, 1, lane, NPAD);
        o2 = wmma_f16(a0, b0, o2); o2 = wmma_f16(a1, b1, o2);
        b0 = load_fragB(Vts, 48, 0, lane, NPAD); b1 = load_fragB(Vts, 48, 1, lane, NPAD);
        o3 = wmma_f16(a0, b0, o3); o3 = wmma_f16(a1, b1, o3);
    }

    // -------- store O --------
    float* ob = out + base;
    float od[4][8];
    #pragma unroll
    for (int r = 0; r < 8; ++r) { od[0][r] = o0[r]; od[1][r] = o1[r]; od[2][r] = o2[r]; od[3][r] = o3[r]; }
    #pragma unroll
    for (int dt = 0; dt < 4; ++dt) {
        int d = dt * 16 + colL;
        #pragma unroll
        for (int r = 0; r < 8; ++r) {
            int n = wave * 16 + rowOff + r;
            if (n < NTOK) ob[n * CDIM + d] = od[dt][r];
        }
    }
}

// ---------------------------------------------------------------------------
extern "C" void kernel_launch(void* const* d_in, const int* in_sizes, int n_in,
                              void* d_out, int out_size, void* d_ws, size_t ws_size,
                              hipStream_t stream) {
    const float* q = (const float*)d_in[0];
    const float* k = (const float*)d_in[1];
    const float* v = (const float*)d_in[2];
    float* table = (float*)((char*)d_ws + WS_TABLE_OFF);  // 169*8 f32
    float* biasF = (float*)((char*)d_ws + WS_BIASF_OFF);  // 8*4*32*32 f32

    pos_mlp_kernel<<<1, 192, 0, stream>>>(
        (const float*)d_in[3],  (const float*)d_in[4],
        (const float*)d_in[5],  (const float*)d_in[6],
        (const float*)d_in[7],  (const float*)d_in[8],
        (const float*)d_in[9],  (const float*)d_in[10],
        (const float*)d_in[11], (const float*)d_in[12],
        (const float*)d_in[13], (const float*)d_in[14],
        (const float*)d_in[15], (const float*)d_in[16],
        table);

    bias_expand_kernel<<<(BIASF_COUNT + 255) / 256, 256, 0, stream>>>(table, biasF);

    int B = in_sizes[0] / (NTOK * CDIM);
    attn_kernel<<<B * HEADS, 128, 0, stream>>>(q, k, v, biasF, (float*)d_out);
}
